// AttentionBlock_45938970198738
// MI455X (gfx1250) — matrix operand order repaired and checked by reading, not compiled, not run on hardware
//
#include <hip/hip_runtime.h>
#include <hip/hip_bf16.h>

typedef _Float16 half16 __attribute__((ext_vector_type(16)));
typedef _Float16 half8  __attribute__((ext_vector_type(8)));
typedef float    float8 __attribute__((ext_vector_type(8)));

#define BATCH 4
#define CH    256
#define HW    4096
#define CONDD 512
#define NG    32

__device__ __forceinline__ float8 wmma_f16(half16 a, half16 b, float8 c) {
  return __builtin_amdgcn_wmma_f32_16x16x32_f16(false, a, false, b, (short)0, c, false, false);
}

// A-fragment (16x32 f16) from row-major [M x ld] at tile (m0, k0).
// Lane L: row = m0 + (L&15); elems 0..7 -> K = k0+8h+e ; 8..15 -> K = k0+16+8h+e (h = L>>4).
__device__ __forceinline__ half16 load_a_frag(const _Float16* p, int ld, int m0, int k0, int lane) {
  const _Float16* r = p + (size_t)(m0 + (lane & 15)) * ld + k0;
  const half8* r8 = (const half8*)r;
  int h = lane >> 4;
  half8 lo = r8[h];
  half8 hi = r8[2 + h];
  half16 a;
#pragma unroll
  for (int e = 0; e < 8; ++e) { a[e] = lo[e]; a[8 + e] = hi[e]; }
  return a;
}

// B-fragment (32x16 f16) from B^T stored row-major [N x ld] at tile (n0, k0).
// Lane L: col N = n0 + (L&15); elems 0..7 -> K = k0+8h+e ; 8..15 -> K = k0+16+8h+e (h = L>>4).
__device__ __forceinline__ half16 load_bT_frag(const _Float16* p, int ld, int n0, int k0, int lane) {
  const half8* r8 = (const half8*)(p + (size_t)(n0 + (lane & 15)) * ld + k0);
  int h = lane >> 4;
  half8 lo = r8[h];
  half8 hi = r8[2 + h];
  half16 b;
#pragma unroll
  for (int e = 0; e < 8; ++e) { b[e] = lo[e]; b[8 + e] = hi[e]; }
  return b;
}

// ---------------- 1. GroupNorm statistics ----------------
__global__ __launch_bounds__(256) void k_gn_stats(const float* __restrict__ x,
                                                  float* __restrict__ mean_ws,
                                                  float* __restrict__ rstd_ws) {
  __shared__ float sh_s[256], sh_q[256];
  int bg = blockIdx.x;             // 0..127  (b*32+g)
  const float* base = x + (size_t)bg * 8 * HW;   // 8 channels per group, contiguous
  const int N = 8 * HW;
  float s = 0.f, q = 0.f;
  for (int i = threadIdx.x; i < N; i += 256) {
    float v = base[i];
    s += v; q += v * v;
  }
  sh_s[threadIdx.x] = s; sh_q[threadIdx.x] = q;
  __syncthreads();
  for (int off = 128; off > 0; off >>= 1) {
    if (threadIdx.x < off) {
      sh_s[threadIdx.x] += sh_s[threadIdx.x + off];
      sh_q[threadIdx.x] += sh_q[threadIdx.x + off];
    }
    __syncthreads();
  }
  if (threadIdx.x == 0) {
    float m = sh_s[0] / (float)N;
    float var = sh_q[0] / (float)N - m * m;
    mean_ws[bg] = m;
    rstd_ws[bg] = rsqrtf(var + 1e-5f);
  }
}

// ---------------- 2. sb = cond @ lin_w + lin_b ----------------
__global__ __launch_bounds__(256) void k_sb(const float* __restrict__ cond,
                                            const float* __restrict__ lin_w,
                                            const float* __restrict__ lin_b,
                                            float* __restrict__ sb) {
  int b = blockIdx.x;
  const float* cb = cond + (size_t)b * CONDD;
#pragma unroll
  for (int rep = 0; rep < 2; ++rep) {
    int j = threadIdx.x + rep * 256;   // 0..511
    float acc = lin_b[j];
    for (int kk = 0; kk < CONDD; ++kk)
      acc += cb[kk] * lin_w[(size_t)kk * 512 + j];
    sb[(size_t)b * 512 + j] = acc;
  }
}

// ---------------- 3. weight prep: transpose + f16 ----------------
__global__ __launch_bounds__(256) void k_prep_w(const float* __restrict__ qkv_w,
                                                const float* __restrict__ proj_w,
                                                _Float16* __restrict__ wqkvT,
                                                _Float16* __restrict__ wprojT) {
  int idx = blockIdx.x * 256 + threadIdx.x;
  const int NQ = 768 * 256;
  if (idx < NQ) {
    int n = idx / 256, kk = idx % 256;
    wqkvT[idx] = (_Float16)qkv_w[(size_t)kk * 768 + n];
  } else {
    int i2 = idx - NQ;               // < 256*256
    int n = i2 / 256, kk = i2 % 256;
    wprojT[i2] = (_Float16)proj_w[(size_t)kk * 256 + n];
  }
}

// ---------------- 4. normalize + affine -> f16 tokens [B,HW,C] ----------------
__global__ __launch_bounds__(256) void k_norm_t(const float* __restrict__ x,
                                                const float* __restrict__ mean_ws,
                                                const float* __restrict__ rstd_ws,
                                                const float* __restrict__ sb,
                                                _Float16* __restrict__ t) {
  int idx = blockIdx.x * 256 + threadIdx.x;     // over B*C*HW
  int b = idx / (CH * HW);
  int rem = idx - b * CH * HW;
  int c = rem / HW;
  int tok = rem - c * HW;
  int g = c >> 3;
  float m = mean_ws[b * NG + g];
  float rs = rstd_ws[b * NG + g];
  float scale = sb[(size_t)b * 512 + c];
  float bias  = sb[(size_t)b * 512 + CH + c];
  float v = (x[idx] - m) * rs * (1.0f + scale) + bias;
  t[((size_t)b * HW + tok) * CH + c] = (_Float16)v;
}

// ---------------- 5. QKV GEMM (WMMA) ----------------
__global__ __launch_bounds__(256, 1) void k_qkv(const _Float16* __restrict__ t,
                                                const _Float16* __restrict__ wqkvT,
                                                const float* __restrict__ qkv_b,
                                                _Float16* __restrict__ q,
                                                _Float16* __restrict__ k,
                                                _Float16* __restrict__ vT) {
  int lane = threadIdx.x & 31, w = threadIdx.x >> 5;
  int b = blockIdx.z;
  int m0 = blockIdx.x * 16;
  int n0 = (blockIdx.y * 8 + w) * 16;           // 0..752
  const _Float16* tb = t + (size_t)b * HW * CH;
  float8 acc = {};
#pragma unroll
  for (int cs = 0; cs < 8; ++cs) {
    half16 A = load_a_frag(tb, CH, m0, 32 * cs, lane);
    half16 B = load_bT_frag(wqkvT, CH, n0, 32 * cs, lane);
    acc = wmma_f16(A, B, acc);
  }
  int col = n0 + (lane & 15);
  int h = lane >> 4;
  float bias = qkv_b[col];
#pragma unroll
  for (int r = 0; r < 8; ++r) {
    int tok = m0 + r + 8 * h;
    float v = acc[r] + bias;
    if (col < CH) {
      q[((size_t)b * HW + tok) * CH + col] = (_Float16)v;
    } else if (col < 2 * CH) {
      k[((size_t)b * HW + tok) * CH + (col - CH)] = (_Float16)v;
    } else {
      vT[((size_t)b * CH + (col - 2 * CH)) * HW + tok] = (_Float16)v;  // V transposed
    }
  }
}

// ---------------- 6. Flash attention (WMMA, online softmax) ----------------
__global__ __launch_bounds__(256, 1) void k_flash(const _Float16* __restrict__ q,
                                                  const _Float16* __restrict__ k,
                                                  const _Float16* __restrict__ vT,
                                                  _Float16* __restrict__ o) {
  __shared__ _Float16 ldsP[8][16 * 32];         // per-wave P scratch (16 queries x 32 keys)
  int lane = threadIdx.x & 31, w = threadIdx.x >> 5;
  int b = blockIdx.y;
  int q0 = (blockIdx.x * 8 + w) * 16;
  const _Float16* qb = q  + (size_t)b * HW * CH;
  const _Float16* kb = k  + (size_t)b * HW * CH;
  const _Float16* vb = vT + (size_t)b * CH * HW;

  half16 Qf[8];
#pragma unroll
  for (int cs = 0; cs < 8; ++cs) Qf[cs] = load_a_frag(qb, CH, q0, 32 * cs, lane);

  float8 Oacc[16];
#pragma unroll
  for (int tnum = 0; tnum < 16; ++tnum) Oacc[tnum] = (float8){};
  float mrow[8], lrow[8];
#pragma unroll
  for (int r = 0; r < 8; ++r) { mrow[r] = -3.0e38f; lrow[r] = 0.0f; }

  const float scl = 0.0625f;                    // 1/sqrt(C)
  int col = lane & 15, h = lane >> 4;
  _Float16* pl = &ldsP[w][0];

  for (int kt = 0; kt < HW; kt += 32) {
    // ---- scores S = Q K^T  (two 16x16 D-tiles, K=256 in 8 WMMA steps each)
    float8 S0 = {}, S1 = {};
#pragma unroll
    for (int cs = 0; cs < 8; ++cs) {
      half16 B0 = load_bT_frag(kb, CH, kt,      32 * cs, lane);
      half16 B1 = load_bT_frag(kb, CH, kt + 16, 32 * cs, lane);
      S0 = wmma_f16(Qf[cs], B0, S0);
      S1 = wmma_f16(Qf[cs], B1, S1);
    }
    // ---- online softmax (rows live in 16-lane halves: reduce with xor 1,2,4,8)
    float P0[8], P1[8], corr[8];
#pragma unroll
    for (int r = 0; r < 8; ++r) {
      float a = S0[r] * scl, bb = S1[r] * scl;
      float mx = fmaxf(a, bb);
#pragma unroll
      for (int off = 1; off < 16; off <<= 1) mx = fmaxf(mx, __shfl_xor(mx, off, 32));
      float mnew = fmaxf(mrow[r], mx);
      corr[r] = __expf(mrow[r] - mnew);
      float p0 = __expf(a - mnew), p1 = __expf(bb - mnew);
      float rs = p0 + p1;
#pragma unroll
      for (int off = 1; off < 16; off <<= 1) rs += __shfl_xor(rs, off, 32);
      lrow[r] = lrow[r] * corr[r] + rs;
      mrow[r] = mnew;
      P0[r] = p0; P1[r] = p1;
    }
#pragma unroll
    for (int tnum = 0; tnum < 16; ++tnum)
#pragma unroll
      for (int r = 0; r < 8; ++r) Oacc[tnum][r] *= corr[r];
    // ---- transpose P (D-layout -> A-layout) via per-wave LDS scratch
#pragma unroll
    for (int r = 0; r < 8; ++r) {
      int row = r + 8 * h;
      pl[row * 32 + col]      = (_Float16)P0[r];
      pl[row * 32 + 16 + col] = (_Float16)P1[r];
    }
    asm volatile("s_wait_dscnt 0" ::: "memory");
    half16 Pa;
    {
      const half8* pr = (const half8*)(pl + (lane & 15) * 32);
      half8 lo = pr[h];
      half8 hi = pr[2 + h];
#pragma unroll
      for (int e = 0; e < 8; ++e) { Pa[e] = lo[e]; Pa[8 + e] = hi[e]; }
    }
    // ---- O += P V   (V^T rows are contiguous K runs)
#pragma unroll
    for (int tnum = 0; tnum < 16; ++tnum) {
      half16 Bv = load_bT_frag(vb, HW, 16 * tnum, kt, lane);
      Oacc[tnum] = wmma_f16(Pa, Bv, Oacc[tnum]);
    }
  }
  // ---- finalize: O /= l ; write f16 [tok][chan]
  float inv[8];
#pragma unroll
  for (int r = 0; r < 8; ++r) inv[r] = 1.0f / lrow[r];
#pragma unroll
  for (int tnum = 0; tnum < 16; ++tnum)
#pragma unroll
    for (int r = 0; r < 8; ++r) {
      int tok = q0 + r + 8 * h;
      o[((size_t)b * HW + tok) * CH + 16 * tnum + col] = (_Float16)(Oacc[tnum][r] * inv[r]);
    }
}

// ---------------- 7. Proj GEMM + bias + residual ----------------
__global__ __launch_bounds__(256, 1) void k_proj(const _Float16* __restrict__ o,
                                                 const _Float16* __restrict__ wprojT,
                                                 const float* __restrict__ proj_b,
                                                 const float* __restrict__ x,
                                                 float* __restrict__ out) {
  int lane = threadIdx.x & 31, w = threadIdx.x >> 5;
  int b = blockIdx.z;
  int m0 = blockIdx.x * 16;
  int n0 = (blockIdx.y * 8 + w) * 16;
  const _Float16* ob = o + (size_t)b * HW * CH;
  float8 acc = {};
#pragma unroll
  for (int cs = 0; cs < 8; ++cs) {
    half16 A = load_a_frag(ob, CH, m0, 32 * cs, lane);
    half16 B = load_bT_frag(wprojT, CH, n0, 32 * cs, lane);
    acc = wmma_f16(A, B, acc);
  }
  int col = n0 + (lane & 15);
  int h = lane >> 4;
  float bias = proj_b[col];
  size_t base = ((size_t)b * CH + col) * HW + m0 + 8 * h;   // [B,C,H,W] layout, 32B aligned
  float8 xv = *(const float8*)(x + base);
  float8 res;
#pragma unroll
  for (int r = 0; r < 8; ++r) res[r] = xv[r] + acc[r] + bias;
  *(float8*)(out + base) = res;
}

extern "C" void kernel_launch(void* const* d_in, const int* in_sizes, int n_in,
                              void* d_out, int out_size, void* d_ws, size_t ws_size,
                              hipStream_t stream) {
  const float* x      = (const float*)d_in[0];
  const float* cond   = (const float*)d_in[1];
  const float* lin_w  = (const float*)d_in[2];
  const float* lin_b  = (const float*)d_in[3];
  const float* qkv_w  = (const float*)d_in[4];
  const float* qkv_b  = (const float*)d_in[5];
  const float* proj_w = (const float*)d_in[6];
  const float* proj_b = (const float*)d_in[7];
  float* out = (float*)d_out;

  char* ws = (char*)d_ws;
  // workspace layout (all offsets >= 32B aligned)
  float*    mean_ws = (float*)(ws + 0);                 // 128 f32
  float*    rstd_ws = (float*)(ws + 1024);              // 128 f32
  float*    sb      = (float*)(ws + 2048);              // 4*512 f32
  _Float16* wqkvT   = (_Float16*)(ws + (64 << 10));     // 768*256 f16 (384 KB)
  _Float16* wprojT  = (_Float16*)(ws + (512 << 10));    // 256*256 f16 (128 KB)
  _Float16* t       = (_Float16*)(ws + (1ull << 20));   // 8 MB
  _Float16* q       = (_Float16*)(ws + (16ull << 20));  // 8 MB
  _Float16* k       = (_Float16*)(ws + (24ull << 20));  // 8 MB
  _Float16* vT      = (_Float16*)(ws + (32ull << 20));  // 8 MB
  _Float16* o       = (_Float16*)(ws + (40ull << 20));  // 8 MB

  k_gn_stats<<<BATCH * NG, 256, 0, stream>>>(x, mean_ws, rstd_ws);
  k_sb<<<BATCH, 256, 0, stream>>>(cond, lin_w, lin_b, sb);
  k_prep_w<<<(768 * 256 + 256 * 256) / 256, 256, 0, stream>>>(qkv_w, proj_w, wqkvT, wprojT);
  k_norm_t<<<(BATCH * CH * HW) / 256, 256, 0, stream>>>(x, mean_ws, rstd_ws, sb, t);
  k_qkv<<<dim3(HW / 16, 6, BATCH), 256, 0, stream>>>(t, wqkvT, qkv_b, q, k, vT);
  k_flash<<<dim3(HW / 128, BATCH), 256, 0, stream>>>(q, k, vT, o);
  k_proj<<<dim3(HW / 16, 2, BATCH), 256, 0, stream>>>(o, wprojT, proj_b, x, out);
}
